// SwinTransformerBlock_19782619365404
// MI455X (gfx1250) — matrix operand order repaired and checked by reading, not compiled, not run on hardware
//
#include <hip/hip_runtime.h>
#include <hip/hip_bf16.h>

// ---------------------------------------------------------------------------
// Swin transformer block for MI455X (gfx1250): all GEMMs via
// v_wmma_f32_16x16x32_f16 (f16 in, f32 accumulate), elementwise in f32.
// All WMMA fragments are loaded as contiguous 128-bit chunks:
//  - A fragments: two b128 loads per lane (row-major A, ISA 7.12.2 layout)
//  - B fragments: weights/activations stored TRANSPOSED so each lane's
//    column-of-B slice is two contiguous 16-byte runs (two b128 loads, no repacking)
// ---------------------------------------------------------------------------

typedef _Float16 half_t;
typedef __attribute__((ext_vector_type(16))) _Float16 v16h;
typedef __attribute__((ext_vector_type(8)))  _Float16 v8h;
typedef __attribute__((ext_vector_type(8)))  float    v8f;
typedef __attribute__((ext_vector_type(4)))  float    v4f;

#define WSZ   7
#define NW    49      // tokens per window
#define NPAD  64      // padded rows (multiple of 16)
#define CCH   128     // channels
#define HEADS 4
#define HD    32      // head dim
#define HH    56
#define WW    56
#define LTOK  3136    // 56*56
#define DFF   512
#define QKN   384     // 3*C

// ---------------- wave32 reductions ----------------
__device__ __forceinline__ float wsum(float v) {
#pragma unroll
  for (int o = 16; o > 0; o >>= 1) v += __shfl_xor(v, o, 32);
  return v;
}
__device__ __forceinline__ float wmaxr(float v) {
#pragma unroll
  for (int o = 16; o > 0; o >>= 1) v = fmaxf(v, __shfl_xor(v, o, 32));
  return v;
}

// ---------------- WMMA fragment loaders (ISA 7.12.2, wave32) ----------------
__device__ __forceinline__ v16h make16(v8h lo, v8h hi) {
  return __builtin_shufflevector(lo, hi, 0, 1, 2, 3, 4, 5, 6, 7,
                                 8, 9, 10, 11, 12, 13, 14, 15);
}
// A: 16x32 f16 tile at A (row-major, leading dim lda, lda % 8 == 0).
// lane l (m = l%16, hs = l/16) holds A[m][hs*8 + 0..7] and A[m][16+hs*8+0..7]:
// two contiguous 16B runs -> two b128 loads.
__device__ __forceinline__ v16h load_a16x32(const half_t* A, int lda, int lane) {
  const int m = lane & 15, hs = lane >> 4;
  const half_t* p = A + m * lda + hs * 8;
  v8h lo = *(const v8h*)(p);
  v8h hi = *(const v8h*)(p + 16);
  return make16(lo, hi);
}
// B: 32x16 f16 tile, sourced from TRANSPOSED storage BT[n][k] (ldb % 8 == 0).
// Caller passes BT pre-offset to (n0, k0). lane l (n = l%16, hs = l/16) needs
// B[k0+hs*8+e][n0+n] and B[k0+16+hs*8+e][n0+n] (e=0..7): two contiguous 16B runs.
__device__ __forceinline__ v16h load_bT32x16(const half_t* BT, int ldb, int lane) {
  const int n = lane & 15, hs = lane >> 4;
  const half_t* p = BT + n * ldb + hs * 8;
  v8h lo = *(const v8h*)(p);
  v8h hi = *(const v8h*)(p + 16);
  return make16(lo, hi);
}
__device__ __forceinline__ v8f wmma_f16(v16h a, v16h b, v8f c) {
  return __builtin_amdgcn_wmma_f32_16x16x32_f16(false, a, false, b, (short)0, c,
                                                false, false);
}

// ---------------- kernel 0: f32 -> f16 TRANSPOSED weight conversion -------
// src is (K, N) row-major f32; dst is (N, K) row-major f16.
__global__ void f2hT_kernel(const float* __restrict__ src, half_t* __restrict__ dst,
                            int K, int N) {
  int i = blockIdx.x * 256 + threadIdx.x;
  if (i < K * N) {
    int k = i / N, n = i % N;
    dst[n * K + k] = (half_t)src[i];
  }
}

// ---------------- kernel 1: LN1 + window partition (pad 49->64) ----------------
__global__ void ln1_window_kernel(const float* __restrict__ x,
                                  const float* __restrict__ gam,
                                  const float* __restrict__ bet,
                                  half_t* __restrict__ xn) {
  const int win = blockIdx.x;
  const int b = win >> 6, wy = (win >> 3) & 7, wx = win & 7;
  const int wave = threadIdx.x >> 5, lane = threadIdx.x & 31;
  for (int m = wave; m < NPAD; m += 8) {
    half_t* dst = xn + ((size_t)win * NPAD + m) * CCH;
    if (m < NW) {
      const int hh = wy * WSZ + m / WSZ;
      const int ww = wx * WSZ + m % WSZ;
      const float* src = x + ((size_t)b * LTOK + hh * WW + ww) * CCH;
      v4f v0 = *(const v4f*)(src + lane * 4);
      float sum = v0.x + v0.y + v0.z + v0.w;
      float sq  = v0.x * v0.x + v0.y * v0.y + v0.z * v0.z + v0.w * v0.w;
      sum = wsum(sum); sq = wsum(sq);
      const float mean = sum * (1.0f / CCH);
      const float var  = sq * (1.0f / CCH) - mean * mean;
      const float rstd = rsqrtf(var + 1e-5f);
#pragma unroll
      for (int j = 0; j < 4; ++j) {
        const int c = lane * 4 + j;
        dst[c] = (half_t)((v0[j] - mean) * rstd * gam[c] + bet[c]);
      }
    } else if (lane < 16) {
      // zero the 128-half pad row: 16 lanes x 8 halfs = exactly one row
      *(v8h*)(dst + lane * 8) = (v8h)0;
    }
  }
}

// ---------------- kernel 2: QKV GEMM (64x128 @ 128x384) per window ----------------
__global__ void qkv_kernel(const half_t* __restrict__ xn,
                           const half_t* __restrict__ wqT,    // (384,128) f16 transposed
                           const float*  __restrict__ bias,   // (384) f32
                           half_t* __restrict__ qb,           // (win,h,64,32) pre-scaled
                           half_t* __restrict__ kb,           // (win,h,64,32) natural
                           half_t* __restrict__ vTb) {        // (win,h,32,64) transposed
  __shared__ __align__(16) half_t sA[NPAD * CCH];
  const int win = blockIdx.x;
  const int tid = threadIdx.x, wave = tid >> 5, lane = tid & 31;
  const v8h* g = (const v8h*)(xn + (size_t)win * NPAD * CCH);
  for (int i = tid; i < NPAD * CCH / 8; i += 256) ((v8h*)sA)[i] = g[i];
  __syncthreads();
  const int hs = lane >> 4, nl = lane & 15;
  for (int t = wave; t < 4 * (QKN / 16); t += 8) {
    const int mt = t / (QKN / 16), nt = t % (QKN / 16);
    const int m0 = mt * 16, n0 = nt * 16;
    // prefetch next tile's weight slab (global_prefetch_b8)
    if (t + 8 < 4 * (QKN / 16))
      __builtin_prefetch(&wqT[(size_t)(((t + 8) % (QKN / 16)) * 16) * CCH], 0, 1);
    v8f acc = {};
#pragma unroll
    for (int kk = 0; kk < 4; ++kk) {
      v16h a  = load_a16x32(&sA[m0 * CCH + kk * 32], CCH, lane);
      v16h bm = load_bT32x16(&wqT[(size_t)n0 * CCH + kk * 32], CCH, lane);
      acc = wmma_f16(a, bm, acc);
    }
    const int j = n0 + nl;
    const int which = j >> 7, h = (j >> 5) & 3, dd = j & 31;
    const float bj = bias[j];
    const size_t base = (size_t)win * HEADS + h;
#pragma unroll
    for (int r = 0; r < 8; ++r) {
      const int m = m0 + r + 8 * hs;
      const float val = acc[r] + bj;
      if (which == 0)      qb[(base * NPAD + m) * HD + dd]  = (half_t)(val * 0.17677669529663689f);
      else if (which == 1) kb[(base * NPAD + m) * HD + dd]  = (half_t)val;
      else                 vTb[(base * HD + dd) * NPAD + m] = (half_t)val;
    }
  }
}

// ---------------- kernel 3: window attention per (window, head) ----------------
__global__ void attn_kernel(const half_t* __restrict__ qb,
                            const half_t* __restrict__ kb,
                            const half_t* __restrict__ vTb,
                            const float*  __restrict__ rel,   // (169, HEADS)
                            half_t* __restrict__ obuf) {      // (win,64,128)
  __shared__ __align__(16) half_t sQ[NPAD * HD];    // 64x32 (Q, row-major)
  __shared__ __align__(16) half_t sK[NPAD * HD];    // 64x32 (K, row-major = (K^T)^T)
  __shared__ __align__(16) half_t sVT[HD * NPAD];   // 32x64 (V^T)
  __shared__ __align__(16) float  sS[NPAD * NPAD];
  __shared__ __align__(16) half_t sP[NPAD * NPAD];
  const int wh = blockIdx.x;
  const int win = wh >> 2, h = wh & 3;
  const int tid = threadIdx.x, wave = tid >> 5, lane = tid & 31;
  const v8h* gq = (const v8h*)(qb  + (size_t)wh * NPAD * HD);
  const v8h* gk = (const v8h*)(kb  + (size_t)wh * NPAD * HD);
  const v8h* gv = (const v8h*)(vTb + (size_t)wh * HD * NPAD);
  for (int i = tid; i < NPAD * HD / 8; i += 128) {
    ((v8h*)sQ)[i] = gq[i]; ((v8h*)sK)[i] = gk[i]; ((v8h*)sVT)[i] = gv[i];
  }
  __syncthreads();
  const int hs = lane >> 4, nl = lane & 15;
  // S = Q @ K^T  (M=64, K=32, N=64) + rel-pos bias + pad mask.
  // B operand is K^T; its transpose is K itself -> sK is the BT storage.
  for (int t = wave; t < 16; t += 4) {
    const int mt = t >> 2, nt = t & 3;
    v16h a  = load_a16x32(&sQ[mt * 16 * HD], HD, lane);
    v16h bm = load_bT32x16(&sK[(nt * 16) * HD], HD, lane);
    v8f acc = {};
    acc = wmma_f16(a, bm, acc);
#pragma unroll
    for (int r = 0; r < 8; ++r) {
      const int m = mt * 16 + r + 8 * hs;
      const int n = nt * 16 + nl;
      float val = acc[r];
      if (n >= NW) val = -1e30f;                // mask pad keys
      else if (m < NW) {
        const int iq = m / WSZ, jq = m % WSZ, ik = n / WSZ, jk = n % WSZ;
        const int idx = (iq - ik + WSZ - 1) * (2 * WSZ - 1) + (jq - jk + WSZ - 1);
        val += rel[idx * HEADS + h];
      }
      sS[m * NPAD + n] = val;
    }
  }
  __syncthreads();
  // row softmax (each wave: 16 rows; each lane: 2 cols)
  for (int m = wave; m < NPAD; m += 4) {
    const float s0 = sS[m * NPAD + lane];
    const float s1 = sS[m * NPAD + lane + 32];
    const float mx = wmaxr(fmaxf(s0, s1));
    const float e0 = __expf(s0 - mx), e1 = __expf(s1 - mx);
    const float inv = 1.0f / wsum(e0 + e1);
    sP[m * NPAD + lane]      = (half_t)(e0 * inv);
    sP[m * NPAD + lane + 32] = (half_t)(e1 * inv);
  }
  __syncthreads();
  // O = P @ V  (M=64, K=64, N=32); BT storage of V is V^T = sVT.
  for (int t = wave; t < 8; t += 4) {
    const int mt = t >> 1, nt = t & 1;
    v8f acc = {};
#pragma unroll
    for (int kk = 0; kk < 2; ++kk) {
      v16h a  = load_a16x32(&sP[mt * 16 * NPAD + kk * 32], NPAD, lane);
      v16h bm = load_bT32x16(&sVT[(nt * 16) * NPAD + kk * 32], NPAD, lane);
      acc = wmma_f16(a, bm, acc);
    }
#pragma unroll
    for (int r = 0; r < 8; ++r) {
      const int m = mt * 16 + r + 8 * hs;
      const int n = nt * 16 + nl;
      obuf[((size_t)win * NPAD + m) * CCH + h * HD + n] = (half_t)acc[r];
    }
  }
}

// ---------------- kernel 4: proj GEMM + window reverse -> d_out (xo) ----------------
__global__ void proj_kernel(const half_t* __restrict__ obuf,
                            const half_t* __restrict__ wpT,   // (128,128) f16 transposed
                            const float*  __restrict__ bias,
                            float* __restrict__ out) {
  __shared__ __align__(16) half_t sO[NPAD * CCH];
  const int win = blockIdx.x;
  const int b = win >> 6, wy = (win >> 3) & 7, wx = win & 7;
  const int tid = threadIdx.x, wave = tid >> 5, lane = tid & 31;
  const v8h* g = (const v8h*)(obuf + (size_t)win * NPAD * CCH);
  for (int i = tid; i < NPAD * CCH / 8; i += 256) ((v8h*)sO)[i] = g[i];
  __syncthreads();
  const int hs = lane >> 4, nl = lane & 15;
  for (int t = wave; t < 32; t += 8) {
    const int mt = t >> 3, nt = t & 7;
    if (t + 8 < 32)
      __builtin_prefetch(&wpT[(size_t)(((t + 8) & 7) * 16) * CCH], 0, 1);
    v8f acc = {};
#pragma unroll
    for (int kk = 0; kk < 4; ++kk) {
      v16h a  = load_a16x32(&sO[mt * 16 * CCH + kk * 32], CCH, lane);
      v16h bm = load_bT32x16(&wpT[(size_t)(nt * 16) * CCH + kk * 32], CCH, lane);
      acc = wmma_f16(a, bm, acc);
    }
    const int n = nt * 16 + nl;
    const float bj = bias[n];
#pragma unroll
    for (int r = 0; r < 8; ++r) {
      const int m = mt * 16 + r + 8 * hs;
      if (m < NW) {
        const int hh = wy * WSZ + m / WSZ, ww = wx * WSZ + m % WSZ;
        out[((size_t)b * LTOK + hh * WW + ww) * CCH + n] = acc[r] + bj;
      }
    }
  }
}

// ---------------- kernel 5: fused LN2 + MLP (GELU) + residual ----------------
// 64-row tile per block; 64x512 f16 hidden kept entirely in LDS (112 KB of
// the 320 KB WGP LDS) to avoid a 100+ MB HBM intermediate.
__global__ void mlp_kernel(float* __restrict__ out,
                           const float* __restrict__ ln2s, const float* __restrict__ ln2b,
                           const half_t* __restrict__ w1T, const float* __restrict__ b1,
                           const half_t* __restrict__ w2T, const float* __restrict__ b2) {
  extern __shared__ __align__(16) char smem[];
  float*  sXo = (float*)smem;                        // 64*128 f32  (32 KB)
  half_t* sX  = (half_t*)(smem + NPAD * CCH * 4);    // 64*128 f16  (16 KB)
  half_t* sF  = (half_t*)(smem + NPAD * CCH * 6);    // 64*512 f16  (64 KB)
  const size_t row0 = (size_t)blockIdx.x * 64;
  const int tid = threadIdx.x, wave = tid >> 5, lane = tid & 31;
  const v4f* g = (const v4f*)(out + row0 * CCH);
  for (int i = tid; i < 64 * CCH / 4; i += 256) ((v4f*)sXo)[i] = g[i];
  __syncthreads();
  // LN2 (one wave per row)
  for (int m = wave; m < 64; m += 8) {
    v4f v0 = *(const v4f*)(&sXo[m * CCH + lane * 4]);
    float sum = v0.x + v0.y + v0.z + v0.w;
    float sq  = v0.x * v0.x + v0.y * v0.y + v0.z * v0.z + v0.w * v0.w;
    sum = wsum(sum); sq = wsum(sq);
    const float mean = sum * (1.0f / CCH);
    const float var  = sq * (1.0f / CCH) - mean * mean;
    const float rstd = rsqrtf(var + 1e-5f);
#pragma unroll
    for (int j = 0; j < 4; ++j) {
      const int c = lane * 4 + j;
      sX[m * CCH + c] = (half_t)((v0[j] - mean) * rstd * ln2s[c] + ln2b[c]);
    }
  }
  __syncthreads();
  const int hs = lane >> 4, nl = lane & 15;
  // GEMM1 (64x128 @ 128x512) + exact GELU
  for (int t = wave; t < 4 * (DFF / 16); t += 8) {
    const int mt = t >> 5, nt = t & 31;
    if (t + 8 < 4 * (DFF / 16))
      __builtin_prefetch(&w1T[(size_t)(((t + 8) & 31) * 16) * CCH], 0, 1);
    v8f acc = {};
#pragma unroll
    for (int kk = 0; kk < 4; ++kk) {
      v16h a  = load_a16x32(&sX[mt * 16 * CCH + kk * 32], CCH, lane);
      v16h bm = load_bT32x16(&w1T[(size_t)(nt * 16) * CCH + kk * 32], CCH, lane);
      acc = wmma_f16(a, bm, acc);
    }
    const int n = nt * 16 + nl;
    const float bj = b1[n];
#pragma unroll
    for (int r = 0; r < 8; ++r) {
      const int m = mt * 16 + r + 8 * hs;
      const float xv = acc[r] + bj;
      sF[m * DFF + n] = (half_t)(0.5f * xv * (1.0f + erff(xv * 0.70710678118654752f)));
    }
  }
  __syncthreads();
  // GEMM2 (64x512 @ 512x128) + bias + residual -> out
  for (int t = wave; t < 32; t += 8) {
    const int mt = t >> 3, nt = t & 7;
    if (t + 8 < 32)
      __builtin_prefetch(&w2T[(size_t)(((t + 8) & 7) * 16) * DFF], 0, 1);
    v8f acc = {};
#pragma unroll
    for (int kk = 0; kk < 16; ++kk) {
      v16h a  = load_a16x32(&sF[mt * 16 * DFF + kk * 32], DFF, lane);
      v16h bm = load_bT32x16(&w2T[(size_t)(nt * 16) * DFF + kk * 32], DFF, lane);
      acc = wmma_f16(a, bm, acc);
    }
    const int n = nt * 16 + nl;
    const float bj = b2[n];
#pragma unroll
    for (int r = 0; r < 8; ++r) {
      const int m = mt * 16 + r + 8 * hs;
      out[(row0 + m) * CCH + n] = sXo[m * CCH + n] + acc[r] + bj;
    }
  }
}

// ---------------------------------------------------------------------------
extern "C" void kernel_launch(void* const* d_in, const int* in_sizes, int n_in,
                              void* d_out, int out_size, void* d_ws, size_t ws_size,
                              hipStream_t stream) {
  (void)n_in; (void)out_size; (void)ws_size;
  const int Bb = in_sizes[0] / (LTOK * CCH);   // batch (32)
  const float* x         = (const float*)d_in[0];
  const float* qkv_w     = (const float*)d_in[3];
  const float* qkv_bias  = (const float*)d_in[4];
  const float* proj_w    = (const float*)d_in[5];
  const float* proj_bias = (const float*)d_in[6];
  const float* rel       = (const float*)d_in[7];
  const float* ln1s      = (const float*)d_in[8];
  const float* ln1b      = (const float*)d_in[9];
  const float* ln2s      = (const float*)d_in[10];
  const float* ln2b      = (const float*)d_in[11];
  const float* mw1       = (const float*)d_in[12];
  const float* mb1       = (const float*)d_in[13];
  const float* mw2       = (const float*)d_in[14];
  const float* mb2       = (const float*)d_in[15];
  float* out = (float*)d_out;

  // workspace layout (byte offsets); obuf aliases xn (dead after qkv_kernel)
  char* ws = (char*)d_ws;
  half_t* qkvwT = (half_t*)(ws + 0);         // (384,128) f16 =  98304 B
  half_t* projwT = (half_t*)(ws + 98304);    // (128,128) f16 =  32768 B
  half_t* w1T   = (half_t*)(ws + 131072);    // (512,128) f16 = 131072 B
  half_t* w2T   = (half_t*)(ws + 262144);    // (128,512) f16 = 131072 B
  size_t off = 393216;
  half_t* xn   = (half_t*)(ws + off);        // nwin*64*128 f16 (also obuf)
  half_t* obuf = xn;
  off += (size_t)Bb * 64 * NPAD * CCH * sizeof(half_t);
  half_t* qb  = (half_t*)(ws + off); off += (size_t)Bb * 64 * HEADS * NPAD * HD * sizeof(half_t);
  half_t* kb  = (half_t*)(ws + off); off += (size_t)Bb * 64 * HEADS * NPAD * HD * sizeof(half_t);
  half_t* vTb = (half_t*)(ws + off);

  const int nwin = Bb * 64;

  f2hT_kernel<<<(CCH * QKN + 255) / 256, 256, 0, stream>>>(qkv_w, qkvwT, CCH, QKN);
  f2hT_kernel<<<(CCH * CCH + 255) / 256, 256, 0, stream>>>(proj_w, projwT, CCH, CCH);
  f2hT_kernel<<<(CCH * DFF + 255) / 256, 256, 0, stream>>>(mw1, w1T, CCH, DFF);
  f2hT_kernel<<<(DFF * CCH + 255) / 256, 256, 0, stream>>>(mw2, w2T, DFF, CCH);

  ln1_window_kernel<<<nwin, 256, 0, stream>>>(x, ln1s, ln1b, xn);
  qkv_kernel<<<nwin, 256, 0, stream>>>(xn, qkvwT, qkv_bias, qb, kb, vTb);
  attn_kernel<<<nwin * HEADS, 128, 0, stream>>>(qb, kb, vTb, rel, obuf);
  proj_kernel<<<nwin, 256, 0, stream>>>(obuf, projwT, proj_bias, out);
  mlp_kernel<<<Bb * NW, 256, 114688, stream>>>(out, ln2s, ln2b, w1T, mb1, w2T, mb2);
}
